// VectorQuantizer2_14061722927604
// MI455X (gfx1250) — matrix operand order repaired and checked by reading, not compiled, not run on hardware
//
#include <hip/hip_runtime.h>
#include <hip/hip_bf16.h>

typedef __attribute__((ext_vector_type(16))) _Float16 v16h;
typedef __attribute__((ext_vector_type(8)))  _Float16 v8h;
typedef __attribute__((ext_vector_type(8)))  float    v8f;

#define NB      16
#define CDIM    256
#define HW      1024          // 32*32
#define NROWS   16384         // NB*HW
#define NE      8192
#define ZQ_SIZE 4194304       // NB*CDIM*HW
#define BTPAD   264           // 256 + 8 halfs padding (16B aligned rows, no bank conflicts)
#define NCH     (NE / 32)     // 256 chunks of 32 codes

#define USE_ASYNC_LDS 1       // gfx1250 GLOBAL_LOAD_ASYNC_TO_LDS path (flip to 0 to revert)

// ---------------------------------------------------------------------------
// Kernel 1: codebook fp32 -> f16, plus ||e||^2 per code (fused block reduce)
// ---------------------------------------------------------------------------
__global__ __launch_bounds__(256) void embed_prep(const float* __restrict__ emb,
                                                  _Float16* __restrict__ emb_h,
                                                  float* __restrict__ e_norm) {
    __shared__ float red[256];
    int e = blockIdx.x, t = threadIdx.x;
    float v = emb[e * CDIM + t];
    emb_h[e * CDIM + t] = (_Float16)v;
    red[t] = v * v;
    __syncthreads();
    #pragma unroll
    for (int s = 128; s > 0; s >>= 1) {
        if (t < s) red[t] += red[t + s];
        __syncthreads();
    }
    if (t == 0) e_norm[e] = red[0];
}

// ---------------------------------------------------------------------------
// Kernel 2: z (B,C,H,W) fp32 -> zh (B*H*W, C) f16 ; LDS-tiled 32x32 transpose
// ---------------------------------------------------------------------------
__global__ __launch_bounds__(256) void transpose_z(const float* __restrict__ z,
                                                   _Float16* __restrict__ zh) {
    __shared__ float tile[32][33];
    int b   = blockIdx.z;
    int hw0 = blockIdx.x * 32;
    int c0  = blockIdx.y * 32;
    int tx = threadIdx.x, ty = threadIdx.y;
    #pragma unroll
    for (int j = 0; j < 4; ++j) {
        int c = c0 + ty + j * 8;
        tile[ty + j * 8][tx] = z[((size_t)(b * CDIM + c)) * HW + hw0 + tx];
    }
    __syncthreads();
    #pragma unroll
    for (int j = 0; j < 4; ++j) {
        int hw = hw0 + ty + j * 8;
        zh[((size_t)(b * HW + hw)) * CDIM + c0 + tx] = (_Float16)tile[tx][ty + j * 8];
    }
}

// ---------------------------------------------------------------------------
// Kernel 3: fused GEMM + argmin.  128 blocks x 8 waves; wave owns 16 rows (A
// panel in 64 VGPRs), sweeps 8192 codes in 32-code chunks (2 WMMA col-tiles).
// Async double-buffered LDS staging hides the next tile's global latency
// behind 16 WMMAs; one workgroup barrier per chunk.
// ---------------------------------------------------------------------------
__global__ __launch_bounds__(256) void vq_argmin(const _Float16* __restrict__ zh,
                                                 const _Float16* __restrict__ emb_h,
                                                 const float* __restrict__ e_norm,
                                                 int* __restrict__ idx_out) {
#if USE_ASYNC_LDS
    __shared__ __align__(16) _Float16 btile[2][32 * BTPAD];
#else
    __shared__ __align__(16) _Float16 btile[1][32 * BTPAD];
#endif

    const int tid  = threadIdx.x;
    const int lane = tid & 31;
    const int wave = tid >> 5;
    const int sel  = lane >> 4;      // 0: lanes 0-15, 1: lanes 16-31
    const int lrow = lane & 15;
    const int m0   = blockIdx.x * 128 + wave * 16;

    // --- load A panel: 16 rows x 256 K as 8 fragments of v16h ---------------
    // ISA A layout (16-bit 16x32): lanes<16 hold K={0..7,16..23}, lanes>=16
    // hold K={8..15,24..31}: two contiguous 16B runs per lane per fragment.
    const _Float16* zr = zh + (size_t)(m0 + lrow) * CDIM;
    v16h a[8];
    #pragma unroll
    for (int kc = 0; kc < 8; ++kc) {
        v8h lo = *(const v8h*)(zr + kc * 32 + sel * 8);
        v8h hi = *(const v8h*)(zr + kc * 32 + 16 + sel * 8);
        a[kc] = __builtin_shufflevector(lo, hi, 0,1,2,3,4,5,6,7,8,9,10,11,12,13,14,15);
    }

    float best[8];
    int   bidx[8];
    #pragma unroll
    for (int r = 0; r < 8; ++r) { best[r] = 3.4e38f; bidx[r] = 0; }

    // staging: 32 codes x 512B = 16KB per chunk; each thread moves 64B
    const int srow = tid >> 3;       // 0..31 code row within chunk
    const int sseg = tid & 7;        // 0..7  64B segment within 512B row

#if USE_ASYNC_LDS
    // Issue 4 async b128 loads: INST_OFFSET applies to both LDS and global
    // addresses, so one LDS base + one global base covers the 64B segment.
    auto issue_tile = [&](int chunk, int buf) {
        const _Float16* src = emb_h + (size_t)(chunk * 32 + srow) * CDIM + sseg * 32;
        unsigned ldsa = (unsigned)(uintptr_t)&btile[buf][srow * BTPAD + sseg * 32];
        asm volatile(
            "global_load_async_to_lds_b128 %0, %1, off\n\t"
            "global_load_async_to_lds_b128 %0, %1, off offset:16\n\t"
            "global_load_async_to_lds_b128 %0, %1, off offset:32\n\t"
            "global_load_async_to_lds_b128 %0, %1, off offset:48"
            :: "v"(ldsa), "v"(src) : "memory");
    };
    issue_tile(0, 0);
#else
    uint4 pf[4];
    {
        const uint4* src = (const uint4*)(emb_h + (size_t)(srow) * CDIM + sseg * 32);
        pf[0] = src[0]; pf[1] = src[1]; pf[2] = src[2]; pf[3] = src[3];
    }
#endif

    for (int chunk = 0; chunk < NCH; ++chunk) {
        const int cb = chunk * 32;

#if USE_ASYNC_LDS
        // my chunk-k async writes complete:
#if __has_builtin(__builtin_amdgcn_s_wait_asynccnt)
        __builtin_amdgcn_s_wait_asynccnt(0);
#else
        asm volatile("s_wait_asynccnt 0x0" ::: "memory");
#endif
        __syncthreads();   // all waves' writes visible; other buffer free
        if (chunk + 1 < NCH) issue_tile(chunk + 1, (chunk + 1) & 1);
        const _Float16* buf = btile[chunk & 1];
#else
        __syncthreads();   // previous chunk's LDS reads done
        {
            uint4* dst = (uint4*)(&btile[0][srow * BTPAD + sseg * 32]);
            dst[0] = pf[0]; dst[1] = pf[1]; dst[2] = pf[2]; dst[3] = pf[3];
        }
        __syncthreads();
        if (chunk + 1 < NCH) {
            const uint4* src = (const uint4*)(emb_h + (size_t)((chunk + 1) * 32 + srow) * CDIM + sseg * 32);
            pf[0] = src[0]; pf[1] = src[1]; pf[2] = src[2]; pf[3] = src[3];
        }
        const _Float16* buf = btile[0];
#endif

        v8f acc0 = {};
        v8f acc1 = {};
        const _Float16* b0 = buf + lrow * BTPAD;
        const _Float16* b1 = buf + (16 + lrow) * BTPAD;
        #pragma unroll
        for (int kc = 0; kc < 8; ++kc) {
            // ISA B layout (16-bit 32x16): lane = column, lanes<16 hold
            // K={0..7,16..23}, lanes>=16 hold K={8..15,24..31}: two 16B runs.
            v8h l0 = *(const v8h*)(b0 + kc * 32 + sel * 8);
            v8h h0 = *(const v8h*)(b0 + kc * 32 + 16 + sel * 8);
            v16h bf0 = __builtin_shufflevector(l0, h0, 0,1,2,3,4,5,6,7,8,9,10,11,12,13,14,15);
            acc0 = __builtin_amdgcn_wmma_f32_16x16x32_f16(
                false, a[kc], false, bf0, (short)0, acc0, false, false);
            v8h l1 = *(const v8h*)(b1 + kc * 32 + sel * 8);
            v8h h1 = *(const v8h*)(b1 + kc * 32 + 16 + sel * 8);
            v16h bf1 = __builtin_shufflevector(l1, h1, 0,1,2,3,4,5,6,7,8,9,10,11,12,13,14,15);
            acc1 = __builtin_amdgcn_wmma_f32_16x16x32_f16(
                false, a[kc], false, bf1, (short)0, acc1, false, false);
        }

        const int   n0  = cb + lrow;          // column tile 0
        const int   n1  = cb + 16 + lrow;     // column tile 1
        const float en0 = e_norm[n0];
        const float en1 = e_norm[n1];
        #pragma unroll
        for (int r = 0; r < 8; ++r) {         // row = m0 + r + 8*sel
            float s0 = en0 - 2.0f * acc0[r];
            if (s0 < best[r]) { best[r] = s0; bidx[r] = n0; }
            float s1 = en1 - 2.0f * acc1[r];
            if (s1 < best[r]) { best[r] = s1; bidx[r] = n1; }
        }
    }

    // --- reduce across the 16 lanes of each half (rows stay separated) ------
    #pragma unroll
    for (int r = 0; r < 8; ++r) {
        float v = best[r];
        int   i = bidx[r];
        #pragma unroll
        for (int off = 1; off < 16; off <<= 1) {
            float ov = __shfl_xor(v, off, 32);
            int   oi = __shfl_xor(i, off, 32);
            if (ov < v || (ov == v && oi < i)) { v = ov; i = oi; }
        }
        if (lrow == 0) idx_out[m0 + r + (sel << 3)] = i;
    }
}

// ---------------------------------------------------------------------------
// Kernel 4: z_q = emb[idx] in NCHW + per-block SSE partials (coalesced on hw)
// ---------------------------------------------------------------------------
__global__ __launch_bounds__(256) void gather_loss(const float* __restrict__ z,
                                                   const float* __restrict__ emb,
                                                   const int* __restrict__ idx,
                                                   float* __restrict__ out,
                                                   float* __restrict__ partials) {
    __shared__ float red[256];
    const int bx = blockIdx.x;
    const int b  = bx >> 8;
    const int c  = bx & 255;
    const int t  = threadIdx.x;
    float local = 0.0f;
    #pragma unroll
    for (int j = 0; j < 4; ++j) {
        int hw = t + j * 256;
        int n  = b * HW + hw;
        int id = idx[n];
        float q  = emb[(size_t)id * CDIM + c];
        size_t o = (size_t)(b * CDIM + c) * HW + hw;
        float zv = z[o];
        out[o] = q;
        float d = q - zv;
        local += d * d;
    }
    red[t] = local;
    __syncthreads();
    #pragma unroll
    for (int s = 128; s > 0; s >>= 1) {
        if (t < s) red[t] += red[t + s];
        __syncthreads();
    }
    if (t == 0) partials[bx] = red[0];
}

// ---------------------------------------------------------------------------
// Kernel 5: deterministic fixed-order sum -> loss = 1.25 * mean(diff^2)
// ---------------------------------------------------------------------------
__global__ __launch_bounds__(256) void finalize_loss(const float* __restrict__ partials,
                                                     float* __restrict__ out) {
    __shared__ float red[256];
    int t = threadIdx.x;
    float local = 0.0f;
    #pragma unroll
    for (int j = 0; j < 16; ++j) local += partials[t + j * 256];
    red[t] = local;
    __syncthreads();
    #pragma unroll
    for (int s = 128; s > 0; s >>= 1) {
        if (t < s) red[t] += red[t + s];
        __syncthreads();
    }
    if (t == 0) out[ZQ_SIZE] = 1.25f * red[0] / (float)ZQ_SIZE;
}

// ---------------------------------------------------------------------------
extern "C" void kernel_launch(void* const* d_in, const int* in_sizes, int n_in,
                              void* d_out, int out_size, void* d_ws, size_t ws_size,
                              hipStream_t stream) {
    const float* z   = (const float*)d_in[0];   // (16,256,32,32) fp32
    const float* emb = (const float*)d_in[1];   // (8192,256) fp32
    float* out = (float*)d_out;                 // 4194304 z_q + 1 loss

    char* ws = (char*)d_ws;
    _Float16* zh     = (_Float16*)(ws);                    //  8 MB
    _Float16* emb_h  = (_Float16*)(ws + 8388608);          //  4 MB
    float*    e_norm = (float*)   (ws + 12582912);         // 32 KB
    int*      idx    = (int*)     (ws + 12615680);         // 64 KB
    float*    parts  = (float*)   (ws + 12681216);         // 16 KB

    embed_prep  <<<NE, 256, 0, stream>>>(emb, emb_h, e_norm);
    transpose_z <<<dim3(32, 8, 16), dim3(32, 8), 0, stream>>>(z, zh);
    vq_argmin   <<<NROWS / 128, 256, 0, stream>>>(zh, emb_h, e_norm, idx);
    gather_loss <<<NB * CDIM, 256, 0, stream>>>(z, emb, idx, out, parts);
    finalize_loss<<<1, 256, 0, stream>>>(parts, out);
}
